// Generator_81277961110201
// MI455X (gfx1250) — matrix operand order repaired and checked by reading, not compiled, not run on hardware
//
#include <hip/hip_runtime.h>
#include <hip/hip_bf16.h>

typedef __attribute__((ext_vector_type(16))) __bf16 v16bf;
typedef __attribute__((ext_vector_type(8)))  float  v8f;

#define HIDDEN   50
#define NFEAT    3
#define TTF      512
#define TGEN     1024
#define BATCH    16384
#define NT       13      /* n-tiles: 208 gate columns (200 used) */
#define KT1      2       /* layer1 K=64  ([x(3) | h1(50)] padded) */
#define KT2      4       /* layer2 K=128 ([h1(50) | h2(50)] padded) */
#define WGSIZE   128     /* 4 waves x 16 batch rows = 64 rows per WG */

/* ---- LDS layout (bytes, 64-aligned) ---- */
#define B1_BYTES (NT*KT1*32*16*2)          /* 26624 */
#define B2_BYTES (NT*KT2*32*16*2)          /* 53248 */
#define B1_OFF   0
#define B2_OFF   (B1_OFF + B1_BYTES)
#define B1P_OFF  (B2_OFF + B2_BYTES)       /* f32[208] permuted bias L1 */
#define B2P_OFF  (B1P_OFF + 832)           /* f32[208] permuted bias L2 */
#define WLIN_OFF (B2P_OFF + 832)           /* f32[150] */
#define BLIN_OFF (WLIN_OFF + 640)          /* f32[3]   */
#define WAVE_OFF (BLIN_OFF + 64)           /* = 82240  */
#define A1_B     4096                      /* [2][KT1*512] bf16 (double buf) */
#define A2_B     8192                      /* [2][KT2*512] bf16 (double buf) */
#define GB_B     1024                      /* [32][8] f32 gate-tile spill */
#define H2_B     3328                      /* [16][52] f32 h2 (for head) */
#define WAVE_BYTES (A1_B + A2_B + GB_B + H2_B)     /* 16640 */
#define SMEM_TOTAL (WAVE_OFF + 4*WAVE_BYTES)       /* 148800 */

/* 16-bit A-matrix 16x32 fragment slot for element (row m, K):
   half = (K&8)?1:0 ; elem = (K&7) + ((K&16)?8:0) ; lane = m + 16*half  */
__device__ __forceinline__ int a_off(int K, int m) {
  int kt   = K >> 5;
  int Kk   = K & 31;
  int half = (Kk >> 3) & 1;
  int e    = (Kk & 7) | ((Kk & 16) >> 1);
  return kt * 512 + ((m + (half << 4)) << 4) + e;
}

/* Branch-free activations on raw v_exp_f32 / v_rcp_f32.
   exp2 saturates to +inf / +0 out of range, so both functions clamp
   naturally at +/-1 (tanh) and 0/1 (sigmoid) without libm range code. */
#define LOG2E 1.4426950408889634f
__device__ __forceinline__ float fast_sig(float v) {
  return __builtin_amdgcn_rcpf(1.0f + __builtin_amdgcn_exp2f(-LOG2E * v));
}
__device__ __forceinline__ float fast_tanh(float v) {
  return 1.0f - 2.0f * __builtin_amdgcn_rcpf(1.0f + __builtin_amdgcn_exp2f((2.0f * LOG2E) * v));
}

__global__ __launch_bounds__(WGSIZE)
void lstm_gen_kernel(const float* __restrict__ x,
                     const float* __restrict__ Wih1, const float* __restrict__ Whh1,
                     const float* __restrict__ bih1, const float* __restrict__ bhh1,
                     const float* __restrict__ Wih2, const float* __restrict__ Whh2,
                     const float* __restrict__ bih2, const float* __restrict__ bhh2,
                     const float* __restrict__ Wlin, const float* __restrict__ blin,
                     float* __restrict__ out)
{
  extern __shared__ char smem[];
  __bf16* B1w = (__bf16*)(smem + B1_OFF);
  __bf16* B2w = (__bf16*)(smem + B2_OFF);
  float*  b1p = (float*)(smem + B1P_OFF);
  float*  b2p = (float*)(smem + B2P_OFF);
  float*  wl  = (float*)(smem + WLIN_OFF);
  float*  bl  = (float*)(smem + BLIN_OFF);

  const int tid  = threadIdx.x;
  const int lane = tid & 31;
  const int wave = tid >> 5;

  /* ---- one-time: convert weights to bf16 in WMMA B-fragment layout.
     Gate columns permuted: n' = 4*j + g (g in {i,f,g,o}) so each 16-col
     N-tile delivers complete gate quads for 4 hidden units. ---- */
  for (int idx = tid; idx < NT*KT1*32*16; idx += WGSIZE) {
    int e  = idx & 15;
    int ln = (idx >> 4) & 31;
    int kt = (idx >> 9) % KT1;
    int nt = idx / (512*KT1);
    int np = nt*16 + (ln & 15);
    int k  = kt*32 + ((ln >> 4) << 3) + (e & 7) + ((e & 8) << 1);   /* B frag: lane=n+16*((k>>3)&1), e=(k&7)+((k&16)?8:0) */
    float v = 0.0f;
    if (np < 4*HIDDEN) {
      int j = np >> 2, g = np & 3, row = g*HIDDEN + j;
      if (k < NFEAT)              v = Wih1[row*NFEAT + k];
      else if (k < NFEAT+HIDDEN)  v = Whh1[row*HIDDEN + (k - NFEAT)];
    }
    B1w[idx] = (__bf16)v;
  }
  for (int idx = tid; idx < NT*KT2*32*16; idx += WGSIZE) {
    int e  = idx & 15;
    int ln = (idx >> 4) & 31;
    int kt = (idx >> 9) % KT2;
    int nt = idx / (512*KT2);
    int np = nt*16 + (ln & 15);
    int k  = kt*32 + ((ln >> 4) << 3) + (e & 7) + ((e & 8) << 1);
    float v = 0.0f;
    if (np < 4*HIDDEN) {
      int j = np >> 2, g = np & 3, row = g*HIDDEN + j;
      if (k < HIDDEN)          v = Wih2[row*HIDDEN + k];
      else if (k < 2*HIDDEN)   v = Whh2[row*HIDDEN + (k - HIDDEN)];
    }
    B2w[idx] = (__bf16)v;
  }
  for (int np = tid; np < 208; np += WGSIZE) {
    float v1 = 0.0f, v2 = 0.0f;
    if (np < 4*HIDDEN) {
      int j = np >> 2, g = np & 3, row = g*HIDDEN + j;
      v1 = bih1[row] + bhh1[row];
      v2 = bih2[row] + bhh2[row];
    }
    b1p[np] = v1; b2p[np] = v2;
  }
  for (int i = tid; i < NFEAT*HIDDEN; i += WGSIZE) wl[i] = Wlin[i];
  if (tid < NFEAT) bl[tid] = blin[tid];

  /* ---- wave-private LDS ---- */
  char*   wbase = smem + WAVE_OFF + wave * WAVE_BYTES;
  __bf16* A1  = (__bf16*)(wbase);                 /* [2][KT1*512] */
  __bf16* A2  = (__bf16*)(wbase + A1_B);          /* [2][KT2*512] */
  float*  gb  = (float*)(wbase + A1_B + A2_B);
  float*  h2f = (float*)(wbase + A1_B + A2_B + GB_B);

  for (int i = lane; i < WAVE_BYTES/4; i += 32) ((unsigned*)wbase)[i] = 0u;
  __syncthreads();

  const int rowbase = blockIdx.x * 64 + wave * 16;
  const int m_el = lane & 15;
  const int jj0  = lane >> 4;
  const int nloc = lane & 15;            /* C-layout column owned by this lane */

  /* cell state lives in registers: the (row, hidden-unit) -> lane mapping is
     identical every step, so lane owns c for its 2 tasks per N-tile. */
  float c1r[NT][2];
  float c2r[NT][2];
  #pragma unroll
  for (int nt = 0; nt < NT; ++nt) {
    c1r[nt][0] = 0.0f; c1r[nt][1] = 0.0f;
    c2r[nt][0] = 0.0f; c2r[nt][1] = 0.0f;
  }

  for (int t = 0; t < TGEN; ++t) {
    const int cur = t & 1, nxt = cur ^ 1;
    __bf16* A1c = A1 + cur * (KT1*512);
    __bf16* A1n = A1 + nxt * (KT1*512);
    __bf16* A2c = A2 + cur * (KT2*512);
    __bf16* A2n = A2 + nxt * (KT2*512);

    /* teacher-forced input -> A1 x-part (K=0..2) */
    if (t < TTF && lane < 16) {
      const float* xp = x + ((long long)t * BATCH + rowbase + lane) * NFEAT;
      #pragma unroll
      for (int f = 0; f < NFEAT; ++f)
        A1c[a_off(f, lane)] = (__bf16)xp[f];
    }

    /* ---- layer 1: gates1 = [x|h1] @ B1, fused elementwise per N-tile ---- */
    v16bf a1f[KT1];
    #pragma unroll
    for (int kt = 0; kt < KT1; ++kt)
      a1f[kt] = *(const v16bf*)(A1c + kt*512 + lane*16);

    #pragma unroll
    for (int nt = 0; nt < NT; ++nt) {
      /* bias pre-baked: every acc VGPR of this lane is column nt*16+nloc */
      float bb = b1p[nt*16 + nloc];
      v8f acc = {bb, bb, bb, bb, bb, bb, bb, bb};
      #pragma unroll
      for (int kt = 0; kt < KT1; ++kt) {
        v16bf bf = *(const v16bf*)(B1w + ((nt*KT1 + kt)*32 + lane)*16);
        acc = __builtin_amdgcn_wmma_f32_16x16x32_bf16(false, a1f[kt], false, bf,
                                                      (short)0, acc, false, false);
      }
      #pragma unroll
      for (int v = 0; v < 8; ++v) gb[lane*8 + v] = acc[v];   /* spill C tile */
      #pragma unroll
      for (int q = 0; q < 2; ++q) {
        if (4*nt + 2*q < HIDDEN) {        /* compile-time guard (HIDDEN even) */
          int jj = jj0 + 2*q;
          int j  = 4*nt + jj;
          int lw0 = (4*jj + ((m_el >> 3) << 4)) * 8 + (m_el & 7);
          float gi = gb[lw0];
          float gf = gb[lw0 + 8];
          float gg = gb[lw0 + 16];
          float go = gb[lw0 + 24];
          float cn = fast_sig(gf) * c1r[nt][q] + fast_sig(gi) * fast_tanh(gg);
          float h  = fast_sig(go) * fast_tanh(cn);
          c1r[nt][q] = cn;
          __bf16 hb = (__bf16)h;
          A2c[a_off(j, m_el)]         = hb;   /* layer-2 input, this step  */
          A1n[a_off(NFEAT + j, m_el)] = hb;   /* recurrent input, next step */
        }
      }
    }

    /* ---- layer 2: gates2 = [h1|h2] @ B2, fused elementwise per N-tile ---- */
    v16bf a2f[KT2];
    #pragma unroll
    for (int kt = 0; kt < KT2; ++kt)
      a2f[kt] = *(const v16bf*)(A2c + kt*512 + lane*16);

    #pragma unroll
    for (int nt = 0; nt < NT; ++nt) {
      float bb = b2p[nt*16 + nloc];
      v8f acc = {bb, bb, bb, bb, bb, bb, bb, bb};
      #pragma unroll
      for (int kt = 0; kt < KT2; ++kt) {
        v16bf bf = *(const v16bf*)(B2w + ((nt*KT2 + kt)*32 + lane)*16);
        acc = __builtin_amdgcn_wmma_f32_16x16x32_bf16(false, a2f[kt], false, bf,
                                                      (short)0, acc, false, false);
      }
      #pragma unroll
      for (int v = 0; v < 8; ++v) gb[lane*8 + v] = acc[v];
      #pragma unroll
      for (int q = 0; q < 2; ++q) {
        if (4*nt + 2*q < HIDDEN) {        /* compile-time guard */
          int jj = jj0 + 2*q;
          int j  = 4*nt + jj;
          int lw0 = (4*jj + ((m_el >> 3) << 4)) * 8 + (m_el & 7);
          float gi = gb[lw0];
          float gf = gb[lw0 + 8];
          float gg = gb[lw0 + 16];
          float go = gb[lw0 + 24];
          float cn = fast_sig(gf) * c2r[nt][q] + fast_sig(gi) * fast_tanh(gg);
          float h  = fast_sig(go) * fast_tanh(cn);
          c2r[nt][q] = cn;
          A2n[a_off(HIDDEN + j, m_el)] = (__bf16)h;  /* h2 -> next step */
          h2f[m_el*52 + j] = h;                      /* f32 for the head */
        }
      }
    }

    /* ---- output head: out = h2 @ Wlin.T + blin; feed back when generating.
       One k-loop, three accumulators: each h2 LDS read feeds 3 FMAs. ---- */
    if (lane < 16) {
      const float* hm = h2f + lane*52;
      float s0 = bl[0], s1 = bl[1], s2 = bl[2];
      for (int k = 0; k < HIDDEN; ++k) {
        float hv = hm[k];
        s0 += wl[0*HIDDEN + k] * hv;
        s1 += wl[1*HIDDEN + k] * hv;
        s2 += wl[2*HIDDEN + k] * hv;
      }
      long long obase = ((long long)t * BATCH + rowbase + lane) * NFEAT;
      out[obase + 0] = s0;
      out[obase + 1] = s1;
      out[obase + 2] = s2;
      if (t + 1 >= TTF && t + 1 < TGEN) {
        A1n[a_off(0, lane)] = (__bf16)s0;   /* autoregressive feedback */
        A1n[a_off(1, lane)] = (__bf16)s1;
        A1n[a_off(2, lane)] = (__bf16)s2;
      }
    }
  }
}

extern "C" void kernel_launch(void* const* d_in, const int* in_sizes, int n_in,
                              void* d_out, int out_size, void* d_ws, size_t ws_size,
                              hipStream_t stream) {
  const float* x    = (const float*)d_in[0];
  const float* Wih1 = (const float*)d_in[1];
  const float* Whh1 = (const float*)d_in[2];
  const float* bih1 = (const float*)d_in[3];
  const float* bhh1 = (const float*)d_in[4];
  const float* Wih2 = (const float*)d_in[5];
  const float* Whh2 = (const float*)d_in[6];
  const float* bih2 = (const float*)d_in[7];
  const float* bhh2 = (const float*)d_in[8];
  const float* Wlin = (const float*)d_in[9];
  const float* blin = (const float*)d_in[10];
  float* out = (float*)d_out;

  dim3 grid(BATCH / 64);   /* 256 WGs x 4 waves x 16 rows */
  dim3 block(WGSIZE);
  lstm_gen_kernel<<<grid, block, SMEM_TOTAL, stream>>>(
      x, Wih1, Whh1, bih1, bhh1, Wih2, Whh2, bih2, bhh2, Wlin, blin, out);
}
